// MasterDrapeModel_7516192768645
// MI455X (gfx1250) — matrix operand order repaired and checked by reading, not compiled, not run on hardware
//
#include <hip/hip_runtime.h>

// ---------------- model dims ----------------
#define NN   131072
#define EE   524288
#define BB_  8
#define LL   6
#define LAT  128
#define HID  256
#define GCK  150   // global-cond valid dim (padded to 160 in ws)

typedef __attribute__((ext_vector_type(16))) __bf16 v16bf;
typedef __attribute__((ext_vector_type(8)))  float  v8f;

constexpr int IM_NODEFEAT = 0, IM_EDGEATTR = 1, IM_EDGECAT = 2, IM_NODECAT = 3;
constexpr int OM_PLAIN = 0, OM_EDGE = 1, OM_NODEFILM = 2;

constexpr int WAVES = 8;              // 256 threads / block, 128 rows / block
constexpr int BSLICE_USH = 16 * 512;  // max B k-slice: 16 n-tiles * 1KB = 16KB (in ushorts)

// ---------------- helpers ----------------
__device__ __forceinline__ unsigned short f2bf(float f) {
  unsigned u = __float_as_uint(f);
  unsigned r = ((u >> 16) & 1u) + 0x7FFFu;   // round-to-nearest-even
  return (unsigned short)((u + r) >> 16);
}

__device__ __forceinline__ void stb4(unsigned short* p, float4 v) {
  union { unsigned short h[4]; uint2 u; } t;
  t.h[0] = f2bf(v.x); t.h[1] = f2bf(v.y); t.h[2] = f2bf(v.z); t.h[3] = f2bf(v.w);
  *(uint2*)p = t.u;
}

__device__ __forceinline__ v8f vzero8() {
  v8f z = {0.f, 0.f, 0.f, 0.f, 0.f, 0.f, 0.f, 0.f};
  return z;
}

union Frag { v16bf v; uint4 q[2]; };

// One GEMM layer: 16 rows/wave x (NT*16) cols, K = KTOT.
// A: per-wave LDS bf16 tile (row stride kstride).
// B: packed k-slices staged cooperatively (256 threads) into double-buffered LDS;
//    B fragments double-buffered in registers so ds-load latency overlaps WMMA.
template<int KTOT, int NT>
__device__ __forceinline__ void gemm_layer(const unsigned short* As, int kstride,
                                           const unsigned short* pw, unsigned short* Bbuf,
                                           int tid, int lane, v8f* acc) {
  constexpr int NKT = KTOT / 32;
  constexpr int NV  = NT / 4;     // uint4 per thread per 16KB(NT=16)/8KB(NT=8) slice
  const int lhalf = lane >> 4;
  const int lrow  = lane & 15;
  // prestage slice 0 (load phase then store phase so loads clause together)
  {
    uint4 tmp[NV];
    const uint4* src = (const uint4*)pw;
#pragma unroll
    for (int i = 0; i < NV; ++i) tmp[i] = src[tid + i * 256];
    uint4* dst = (uint4*)Bbuf;
#pragma unroll
    for (int i = 0; i < NV; ++i) dst[tid + i * 256] = tmp[i];
  }
  __syncthreads();
  for (int kt = 0; kt < NKT; ++kt) {
    uint4 tmp[NV];
    const bool have = (kt + 1) < NKT;
    if (have) {
      const uint4* src = (const uint4*)(pw + (size_t)(kt + 1) * NT * 512);
#pragma unroll
      for (int i = 0; i < NV; ++i) tmp[i] = src[tid + i * 256];
    }
    if ((kt + 2) < NKT)
      __builtin_prefetch(pw + (size_t)(kt + 2) * NT * 512 + tid * 8, 0, 1);  // -> global_prefetch
    // compute on current slice
    const unsigned short* Bcur = Bbuf + (size_t)(kt & 1) * BSLICE_USH;
    Frag a;
    const unsigned short* rowp = As + lrow * kstride + kt * 32 + lhalf * 8;
    a.q[0] = *(const uint4*)(rowp);
    a.q[1] = *(const uint4*)(rowp + 16);
    Frag bcur, bnext;
    {
      const uint4* bp = (const uint4*)(Bcur + lane * 16);
      bcur.q[0] = bp[0];
      bcur.q[1] = bp[1];
    }
#pragma unroll
    for (int nt = 0; nt < NT; ++nt) {
      if (nt + 1 < NT) {   // prefetch next B fragment before issuing this WMMA
        const uint4* bp = (const uint4*)(Bcur + (nt + 1) * 512 + lane * 16);
        bnext.q[0] = bp[0];
        bnext.q[1] = bp[1];
      }
      acc[nt] = __builtin_amdgcn_wmma_f32_16x16x32_bf16(
          false, a.v, false, bcur.v, (short)0, acc[nt], false, false);
      if (nt + 1 < NT) bcur = bnext;
    }
    if (have) {
      uint4* dst = (uint4*)(Bbuf + (size_t)((kt + 1) & 1) * BSLICE_USH);
#pragma unroll
      for (int i = 0; i < NV; ++i) dst[tid + i * 256] = tmp[i];
    }
    __syncthreads();  // B handoff: everyone done reading cur, next slice visible
  }
}

// bias + LayerNorm + ReLU over NT*16 cols, write bf16 A-tile (stride HID) to LDS.
template<int NT>
__device__ __forceinline__ void ln_relu_store(const v8f* acc, const float* bias,
                                              const float* g, const float* bb,
                                              unsigned short* As2, int lane) {
  const int lhalf = lane >> 4, c0 = lane & 15;
  float s[8], ss[8];
#pragma unroll
  for (int r = 0; r < 8; ++r) { s[r] = 0.f; ss[r] = 0.f; }
#pragma unroll
  for (int nt = 0; nt < NT; ++nt) {
    float bc = bias[nt * 16 + c0];
#pragma unroll
    for (int r = 0; r < 8; ++r) { float v = acc[nt][r] + bc; s[r] += v; ss[r] += v * v; }
  }
#pragma unroll
  for (int r = 0; r < 8; ++r) {
#pragma unroll
    for (int m = 1; m < 16; m <<= 1) {
      s[r]  += __shfl_xor(s[r],  m, 32);
      ss[r] += __shfl_xor(ss[r], m, 32);
    }
  }
  const float inv = 1.0f / (16.0f * NT);
  float mean[8], rstd[8];
#pragma unroll
  for (int r = 0; r < 8; ++r) {
    mean[r] = s[r] * inv;
    float var = ss[r] * inv - mean[r] * mean[r];
    rstd[r] = rsqrtf(var + 1e-5f);
  }
#pragma unroll
  for (int nt = 0; nt < NT; ++nt) {
    int col = nt * 16 + c0;
    float bc = bias[col], gc = g[col], bbc = bb[col];
#pragma unroll
    for (int r = 0; r < 8; ++r) {
      float v = acc[nt][r] + bc;
      v = (v - mean[r]) * rstd[r] * gc + bbc;
      v = fmaxf(v, 0.f);
      As2[(r + lhalf * 8) * HID + col] = f2bf(v);
    }
  }
}

// ---------------- fused 3-layer MLP ----------------
struct MlpArgs {
  const float* a0; const float* a1; const float* a2;
  const int* srcI; const int* dstI; const int* batch;
  const unsigned short *pw1, *pw2, *pw3;
  const float *b1, *g1, *bb1, *b2, *g2, *bb2, *b3;
  float* outbuf;      // x or e (residual read for OM_EDGE / OM_NODEFILM)
  float* aggbuf;      // OM_EDGE
  const float* film;  // OM_NODEFILM: [8][256] gamma||beta
  const float *lng, *lnb;
};

template<int KPAD1, int IM, int OM>
__global__ __launch_bounds__(256) void mlp_fused(MlpArgs A) {
  constexpr int KMAX = (KPAD1 > HID) ? KPAD1 : HID;
  extern __shared__ __align__(16) char smem_raw[];
  unsigned short* Asall = (unsigned short*)smem_raw;
  unsigned short* Bbuf  = (unsigned short*)(smem_raw + (size_t)WAVES * 16 * KMAX * 2);
  const int tid  = threadIdx.x;
  const int lane = tid & 31;
  const int wave = tid >> 5;
  unsigned short* As = Asall + (size_t)wave * 16 * KMAX;
  const int row0 = (blockIdx.x * WAVES + wave) * 16;

  // ---- stage layer-1 A tile (fp32 -> bf16 into LDS, per wave) ----
  if (IM == IM_EDGECAT) {
    for (int rr = 0; rr < 16; ++rr) {
      int er = row0 + rr;
      int sidx = A.srcI[er], didx = A.dstI[er];
      float4 vs = ((const float4*)(A.a0 + (size_t)sidx * LAT))[lane];
      float4 vd = ((const float4*)(A.a0 + (size_t)didx * LAT))[lane];
      float4 ve = ((const float4*)(A.a1 + (size_t)er   * LAT))[lane];
      stb4(As + rr * KPAD1 + lane * 4,            vs);
      stb4(As + rr * KPAD1 + LAT + lane * 4,      vd);
      stb4(As + rr * KPAD1 + 2 * LAT + lane * 4,  ve);
    }
  } else if (IM == IM_NODECAT) {
    for (int rr = 0; rr < 16; ++rr) {
      int nr = row0 + rr;
      float4 vx = ((const float4*)(A.a0 + (size_t)nr * LAT))[lane];
      float4 vg = ((const float4*)(A.a1 + (size_t)nr * LAT))[lane];
      stb4(As + rr * KPAD1 + lane * 4,       vx);
      stb4(As + rr * KPAD1 + LAT + lane * 4, vg);
    }
  } else if (IM == IM_NODEFEAT) {
    if (lane < 16) {
      int nr = row0 + lane;
      unsigned short* rp = As + lane * KPAD1;
      rp[0] = f2bf(A.a0[nr * 3 + 0]); rp[1] = f2bf(A.a0[nr * 3 + 1]); rp[2] = f2bf(A.a0[nr * 3 + 2]);
      rp[3] = f2bf(A.a1[nr * 2 + 0]); rp[4] = f2bf(A.a1[nr * 2 + 1]);
      rp[5] = f2bf(A.a2[nr * 3 + 0]); rp[6] = f2bf(A.a2[nr * 3 + 1]); rp[7] = f2bf(A.a2[nr * 3 + 2]);
#pragma unroll
      for (int j = 8; j < KPAD1; ++j) rp[j] = 0;
    }
  } else {  // IM_EDGEATTR
    if (lane < 16) {
      int er = row0 + lane;
      unsigned short* rp = As + lane * KPAD1;
      rp[0] = f2bf(A.a0[er * 4 + 0]); rp[1] = f2bf(A.a0[er * 4 + 1]);
      rp[2] = f2bf(A.a0[er * 4 + 2]); rp[3] = f2bf(A.a0[er * 4 + 3]);
#pragma unroll
      for (int j = 4; j < KPAD1; ++j) rp[j] = 0;
    }
  }

  // ---- layer 1: KPAD1 -> 256, LN+ReLU ----
  v8f acc[16];
#pragma unroll
  for (int nt = 0; nt < 16; ++nt) acc[nt] = vzero8();
  gemm_layer<KPAD1, 16>(As, KPAD1, A.pw1, Bbuf, tid, lane, acc);
  ln_relu_store<16>(acc, A.b1, A.g1, A.bb1, As, lane);

  // ---- layer 2: 256 -> 256, LN+ReLU ----
#pragma unroll
  for (int nt = 0; nt < 16; ++nt) acc[nt] = vzero8();
  gemm_layer<HID, 16>(As, HID, A.pw2, Bbuf, tid, lane, acc);
  ln_relu_store<16>(acc, A.b2, A.g2, A.bb2, As, lane);

  // ---- layer 3: 256 -> 128 ----
  v8f acc3[8];
#pragma unroll
  for (int nt = 0; nt < 8; ++nt) acc3[nt] = vzero8();
  gemm_layer<HID, 8>(As, HID, A.pw3, Bbuf, tid, lane, acc3);

  const int lhalf = lane >> 4, c0 = lane & 15;
  if (OM == OM_PLAIN) {
#pragma unroll
    for (int nt = 0; nt < 8; ++nt) {
      int col = nt * 16 + c0;
      float bc = A.b3[col];
#pragma unroll
      for (int r = 0; r < 8; ++r) {
        int grow = row0 + r + lhalf * 8;
        A.outbuf[(size_t)grow * LAT + col] = acc3[nt][r] + bc;
      }
    }
  } else if (OM == OM_EDGE) {
#pragma unroll
    for (int nt = 0; nt < 8; ++nt) {
      int col = nt * 16 + c0;
      float bc = A.b3[col];
#pragma unroll
      for (int r = 0; r < 8; ++r) {
        int grow = row0 + r + lhalf * 8;
        size_t off = (size_t)grow * LAT + col;
        float v = A.outbuf[off] + acc3[nt][r] + bc;   // e = e + mlp(...)
        A.outbuf[off] = v;
        atomicAdd(&A.aggbuf[(size_t)A.dstI[grow] * LAT + col], v);
      }
    }
  } else {  // OM_NODEFILM: nu = LN(out), x += gamma*nu + beta
    float s[8], ss[8];
#pragma unroll
    for (int r = 0; r < 8; ++r) { s[r] = 0.f; ss[r] = 0.f; }
#pragma unroll
    for (int nt = 0; nt < 8; ++nt) {
      float bc = A.b3[nt * 16 + c0];
#pragma unroll
      for (int r = 0; r < 8; ++r) { float v = acc3[nt][r] + bc; s[r] += v; ss[r] += v * v; }
    }
#pragma unroll
    for (int r = 0; r < 8; ++r) {
#pragma unroll
      for (int m = 1; m < 16; m <<= 1) {
        s[r]  += __shfl_xor(s[r],  m, 32);
        ss[r] += __shfl_xor(ss[r], m, 32);
      }
    }
    float mean[8], rstd[8];
#pragma unroll
    for (int r = 0; r < 8; ++r) {
      mean[r] = s[r] * (1.0f / 128.0f);
      float var = ss[r] * (1.0f / 128.0f) - mean[r] * mean[r];
      rstd[r] = rsqrtf(var + 1e-5f);
    }
#pragma unroll
    for (int nt = 0; nt < 8; ++nt) {
      int col = nt * 16 + c0;
      float bc = A.b3[col], gc = A.lng[col], bbc = A.lnb[col];
#pragma unroll
      for (int r = 0; r < 8; ++r) {
        float v = acc3[nt][r] + bc;
        float nu = (v - mean[r]) * rstd[r] * gc + bbc;
        int grow = row0 + r + lhalf * 8;
        int bi = A.batch[grow];
        float gamma = A.film[bi * 2 * LAT + col];
        float beta  = A.film[bi * 2 * LAT + LAT + col];
        size_t off = (size_t)grow * LAT + col;
        A.outbuf[off] = A.outbuf[off] + gamma * nu + beta;
      }
    }
  }
}

// ---------------- weight packer: f32 row-major (din x dout) -> bf16 B-fragment tiles ----------------
__global__ void pack_w(const float* W, unsigned int* out, int din, int dout, int ndw) {
  int t = blockIdx.x * 256 + threadIdx.x;
  if (t >= ndw) return;
  int v    = t & 7;
  int lane = (t >> 3) & 31;
  int tile = t >> 8;
  int NT = dout >> 4;
  int nt = tile % NT;
  int kt = tile / NT;
  int K   = kt * 32 + ((lane >> 4) << 3) + ((v >> 2) << 3) + 2 * v;
  int col = nt * 16 + (lane & 15);
  float a = (K     < din) ? W[(size_t)K       * dout + col] : 0.f;
  float b = (K + 1 < din) ? W[(size_t)(K + 1) * dout + col] : 0.f;
  out[t] = (unsigned int)f2bf(a) | ((unsigned int)f2bf(b) << 16);
}

// ---------------- small dense stages ----------------
struct StyleArgs {
  const float *sf, *pW1, *pb1, *plg, *plb, *pW2, *pb2;
  const float *smpl, *phys, *sz, *fabW, *fabb;
  float *styleEmb, *gcond, *fabOut;
};

__global__ void style_kernel(StyleArgs S) {
  __shared__ float h[BB_ * 256];
  __shared__ float mv[BB_ * 2];
  __shared__ float emb[BB_ * 128];
  int tid = threadIdx.x;  // 256 threads
  for (int o = tid; o < BB_ * 256; o += 256) {
    int b = o >> 8, c = o & 255;
    float acc = S.pb1[c];
    for (int k = 0; k < 384; ++k) acc += S.sf[b * 384 + k] * S.pW1[k * 256 + c];
    h[o] = 0.5f * acc * (1.f + erff(acc * 0.7071067811865476f));  // exact gelu
  }
  __syncthreads();
  if (tid < BB_) {
    float s = 0.f, ss = 0.f;
    for (int c = 0; c < 256; ++c) { float v = h[tid * 256 + c]; s += v; ss += v * v; }
    float m = s * (1.f / 256.f);
    float var = ss * (1.f / 256.f) - m * m;
    mv[tid * 2] = m; mv[tid * 2 + 1] = rsqrtf(var + 1e-5f);
  }
  __syncthreads();
  for (int o = tid; o < BB_ * 128; o += 256) {
    int b = o >> 7, c = o & 127;
    float m = mv[b * 2], rs = mv[b * 2 + 1];
    float acc = S.pb2[c];
    for (int k = 0; k < 256; ++k) {
      float v = (h[b * 256 + k] - m) * rs * S.plg[k] + S.plb[k];
      acc += v * S.pW2[k * 128 + c];
    }
    emb[o] = acc; S.styleEmb[o] = acc;
  }
  __syncthreads();
  for (int o = tid; o < BB_ * 160; o += 256) {
    int b = o / 160, j = o % 160;
    float v;
    if      (j < 128) v = emb[b * 128 + j];
    else if (j < 138) v = S.smpl[b * 10 + j - 128];
    else if (j < 148) v = S.phys[b * 10 + j - 138];
    else if (j < 150) v = S.sz[b * 2 + j - 148];
    else              v = 0.f;
    S.gcond[o] = v;
  }
  if (tid < 48) {
    int b = tid / 6, c = tid % 6;
    float acc = S.fabb[c];
    for (int k = 0; k < 128; ++k) acc += emb[b * 128 + k] * S.fabW[k * 6 + c];
    S.fabOut[tid] = acc;
  }
}

struct FilmArgs { const float* fW[LL]; const float* fb[LL]; const float* gc; float* out; };

__global__ void film_kernel(FilmArgs F) {
  int t = blockIdx.x * 256 + threadIdx.x;
  if (t >= LL * BB_ * 2 * LAT) return;
  int l = t / (BB_ * 2 * LAT);
  int rem = t % (BB_ * 2 * LAT);
  int b = rem / (2 * LAT), c = rem % (2 * LAT);
  const float* fW = F.fW[l];
  float acc = F.fb[l][c];
  for (int k = 0; k < GCK; ++k) acc += F.gc[b * 160 + k] * fW[k * 2 * LAT + c];
  F.out[t] = acc;
}

__global__ void decode_kernel(const float* x, const float* dW, const float* db, float* out) {
  int n = blockIdx.x * 256 + threadIdx.x;
  if (n >= NN) return;
  float a0 = db[0], a1 = db[1], a2 = db[2];
  const float* xr = x + (size_t)n * LAT;
  for (int k = 0; k < LAT; ++k) {
    float v = xr[k];
    a0 += v * dW[k * 3 + 0];
    a1 += v * dW[k * 3 + 1];
    a2 += v * dW[k * 3 + 2];
  }
  out[n * 3 + 0] = a0; out[n * 3 + 1] = a1; out[n * 3 + 2] = a2;
}

// ---------------- host orchestration ----------------
extern "C" void kernel_launch(void* const* d_in, const int* in_sizes, int n_in,
                              void* d_out, int out_size, void* d_ws, size_t ws_size,
                              hipStream_t stream) {
  (void)in_sizes; (void)n_in; (void)out_size; (void)ws_size;
  const float* pos  = (const float*)d_in[0];
  const float* uvs  = (const float*)d_in[1];
  const float* nor  = (const float*)d_in[2];
  const float* eatt = (const float*)d_in[3];
  const int* eidx   = (const int*)d_in[8];
  const int* batch  = (const int*)d_in[9];
  const int* srcI = eidx;
  const int* dstI = eidx + EE;
  auto P = [&](int i) { return (const float*)d_in[i]; };

  // param leaf indices (jax dict flatten: sorted keys; lists in order)
  // 10..141: blocks[l] = edge_mlp[0..9], ln_b, ln_g, node_mlp[0..9]
  const int DEC_W = 142, DEC_B = 143, EDGE_ENC = 144, FAB_W = 154, FAB_B = 155;
  const int FILM0 = 156, NODE_ENC = 168, PROJ = 178;

  // workspace bump allocator
  char* wsb = (char*)d_ws;
  size_t off = 0;
  auto alloc = [&](size_t bytes) {
    void* p = wsb + off;
    off = (off + bytes + 255) & ~(size_t)255;
    return p;
  };
  float* xbuf    = (float*)alloc((size_t)NN * LAT * 4);
  float* ebuf    = (float*)alloc((size_t)EE * LAT * 4);
  float* aggbuf  = (float*)alloc((size_t)NN * LAT * 4);
  float* styleEmb= (float*)alloc((size_t)BB_ * 128 * 4);
  float* gcond   = (float*)alloc((size_t)BB_ * 160 * 4);
  float* filmOut = (float*)alloc((size_t)LL * BB_ * 256 * 4);
  auto allocPW = [&](int kpad, int dout) { return (unsigned short*)alloc((size_t)kpad * dout * 2); };

  unsigned short* pwNE[3]  = { allocPW(32, 256),  allocPW(256, 256), allocPW(256, 128) };
  unsigned short* pwEEn[3] = { allocPW(32, 256),  allocPW(256, 256), allocPW(256, 128) };
  unsigned short* pwBE[LL][3];
  unsigned short* pwBN[LL][3];
  for (int l = 0; l < LL; ++l) {
    pwBE[l][0] = allocPW(384, 256); pwBE[l][1] = allocPW(256, 256); pwBE[l][2] = allocPW(256, 128);
    pwBN[l][0] = allocPW(256, 256); pwBN[l][1] = allocPW(256, 256); pwBN[l][2] = allocPW(256, 128);
  }

  auto packW = [&](const float* W, unsigned short* dst, int din, int kpad, int dout) {
    int ndw = kpad * dout / 2;
    pack_w<<<(ndw + 255) / 256, 256, 0, stream>>>(W, (unsigned int*)dst, din, dout, ndw);
  };

  // pack all GEMM weights to bf16 fragment layout
  packW(P(NODE_ENC + 0), pwNE[0],  8,   32,  256);
  packW(P(NODE_ENC + 4), pwNE[1],  256, 256, 256);
  packW(P(NODE_ENC + 8), pwNE[2],  256, 256, 128);
  packW(P(EDGE_ENC + 0), pwEEn[0], 4,   32,  256);
  packW(P(EDGE_ENC + 4), pwEEn[1], 256, 256, 256);
  packW(P(EDGE_ENC + 8), pwEEn[2], 256, 256, 128);
  for (int l = 0; l < LL; ++l) {
    int base = 10 + 22 * l;
    packW(P(base + 0),  pwBE[l][0], 384, 384, 256);
    packW(P(base + 4),  pwBE[l][1], 256, 256, 256);
    packW(P(base + 8),  pwBE[l][2], 256, 256, 128);
    packW(P(base + 12), pwBN[l][0], 256, 256, 256);
    packW(P(base + 16), pwBN[l][1], 256, 256, 256);
    packW(P(base + 20), pwBN[l][2], 256, 256, 128);
  }

  // dynamic LDS sizes: WAVES A-tiles + 32KB double-buffered B
  const size_t shEdge = (size_t)WAVES * 16 * 384 * 2 + 2 * (size_t)BSLICE_USH * 2;  // 128KB
  const size_t shStd  = (size_t)WAVES * 16 * 256 * 2 + 2 * (size_t)BSLICE_USH * 2;  //  96KB

  // style projection + global_cond + fabric logits
  StyleArgs sa{};
  sa.sf = P(4); sa.pW1 = P(PROJ + 0); sa.pb1 = P(PROJ + 1); sa.plg = P(PROJ + 2);
  sa.plb = P(PROJ + 3); sa.pW2 = P(PROJ + 4); sa.pb2 = P(PROJ + 5);
  sa.smpl = P(5); sa.phys = P(6); sa.sz = P(7);
  sa.fabW = P(FAB_W); sa.fabb = P(FAB_B);
  sa.styleEmb = styleEmb; sa.gcond = gcond;
  sa.fabOut = (float*)d_out + (size_t)NN * 3;
  style_kernel<<<1, 256, 0, stream>>>(sa);

  // FiLM tables for all 6 layers
  FilmArgs fa{};
  for (int l = 0; l < LL; ++l) { fa.fW[l] = P(FILM0 + 2 * l); fa.fb[l] = P(FILM0 + 2 * l + 1); }
  fa.gc = gcond; fa.out = filmOut;
  film_kernel<<<(LL * BB_ * 2 * LAT + 255) / 256, 256, 0, stream>>>(fa);

  // node encoder: x = mlp(pos||uvs||normals)
  {
    MlpArgs a{};
    a.a0 = pos; a.a1 = uvs; a.a2 = nor;
    a.pw1 = pwNE[0]; a.pw2 = pwNE[1]; a.pw3 = pwNE[2];
    a.b1 = P(NODE_ENC + 1); a.g1 = P(NODE_ENC + 2); a.bb1 = P(NODE_ENC + 3);
    a.b2 = P(NODE_ENC + 5); a.g2 = P(NODE_ENC + 6); a.bb2 = P(NODE_ENC + 7);
    a.b3 = P(NODE_ENC + 9);
    a.outbuf = xbuf;
    mlp_fused<32, IM_NODEFEAT, OM_PLAIN><<<NN / 128, 256, shStd, stream>>>(a);
  }
  // edge encoder: e = mlp(edge_attr)
  {
    MlpArgs a{};
    a.a0 = eatt;
    a.pw1 = pwEEn[0]; a.pw2 = pwEEn[1]; a.pw3 = pwEEn[2];
    a.b1 = P(EDGE_ENC + 1); a.g1 = P(EDGE_ENC + 2); a.bb1 = P(EDGE_ENC + 3);
    a.b2 = P(EDGE_ENC + 5); a.g2 = P(EDGE_ENC + 6); a.bb2 = P(EDGE_ENC + 7);
    a.b3 = P(EDGE_ENC + 9);
    a.outbuf = ebuf;
    mlp_fused<32, IM_EDGEATTR, OM_PLAIN><<<EE / 128, 256, shStd, stream>>>(a);
  }

  // message-passing layers
  for (int l = 0; l < LL; ++l) {
    int base = 10 + 22 * l;
    hipMemsetAsync(aggbuf, 0, (size_t)NN * LAT * 4, stream);
    {
      MlpArgs a{};
      a.a0 = xbuf; a.a1 = ebuf; a.srcI = srcI; a.dstI = dstI;
      a.pw1 = pwBE[l][0]; a.pw2 = pwBE[l][1]; a.pw3 = pwBE[l][2];
      a.b1 = P(base + 1); a.g1 = P(base + 2); a.bb1 = P(base + 3);
      a.b2 = P(base + 5); a.g2 = P(base + 6); a.bb2 = P(base + 7);
      a.b3 = P(base + 9);
      a.outbuf = ebuf; a.aggbuf = aggbuf;
      mlp_fused<384, IM_EDGECAT, OM_EDGE><<<EE / 128, 256, shEdge, stream>>>(a);
    }
    {
      MlpArgs a{};
      a.a0 = xbuf; a.a1 = aggbuf; a.batch = batch;
      a.pw1 = pwBN[l][0]; a.pw2 = pwBN[l][1]; a.pw3 = pwBN[l][2];
      a.b1 = P(base + 13); a.g1 = P(base + 14); a.bb1 = P(base + 15);
      a.b2 = P(base + 17); a.g2 = P(base + 18); a.bb2 = P(base + 19);
      a.b3 = P(base + 21);
      a.lnb = P(base + 10); a.lng = P(base + 11);
      a.outbuf = xbuf; a.film = filmOut + (size_t)l * BB_ * 256;
      mlp_fused<256, IM_NODECAT, OM_NODEFILM><<<NN / 128, 256, shStd, stream>>>(a);
    }
  }

  // decoder: predicted_delta = x @ dec_W + dec_b
  decode_kernel<<<NN / 256, 256, 0, stream>>>(xbuf, P(DEC_W), P(DEC_B), (float*)d_out);
}
